// CA_2568390443063
// MI455X (gfx1250) — matrix operand order repaired and checked by reading, not compiled, not run on hardware
//
#include <hip/hip_runtime.h>
#include <hip/hip_bf16.h>

typedef __bf16 bf16_t;
typedef __attribute__((ext_vector_type(16))) __bf16 v16bf;
typedef __attribute__((ext_vector_type(8)))  float  v8f;

#define B_  8
#define C_  128
#define H_  256
#define W_  256
#define N_  (H_ * W_)       // 65536
#define HP_ 64
#define NP_ (HP_ * HP_)     // 4096
#define EPS_ 1e-5f

// ---------------------------------------------------------------------------
// Kernel 1/2: depthwise 3x3 stride-2 pad-1 conv + folded BN + PReLU
// one thread per output pixel; block = one output row
// ---------------------------------------------------------------------------
template <int HIN>
__global__ void dwconv_bn_prelu(const float* __restrict__ x,
                                const float* __restrict__ w9,
                                const float* __restrict__ gamma,
                                const float* __restrict__ beta,
                                const float* __restrict__ mean,
                                const float* __restrict__ var,
                                const float* __restrict__ alpha,
                                float* __restrict__ out) {
    constexpr int HOUT = HIN / 2;
    const int ow = threadIdx.x;        // 0..HOUT-1
    const int oh = blockIdx.x;         // 0..HOUT-1
    const int c  = blockIdx.y;         // 0..C_-1
    const int b  = blockIdx.z;         // 0..B_-1

    const float* xi = x + ((size_t)(b * C_ + c)) * HIN * HIN;
    const float* wc = w9 + c * 9;
    float k0 = wc[0], k1 = wc[1], k2 = wc[2];
    float k3 = wc[3], k4 = wc[4], k5 = wc[5];
    float k6 = wc[6], k7 = wc[7], k8 = wc[8];

    const int ih0 = 2 * oh - 1;
    const int iw0 = 2 * ow - 1;
    float acc = 0.f;
#pragma unroll
    for (int r = 0; r < 3; ++r) {
        const int ih = ih0 + r;
        if ((unsigned)ih < (unsigned)HIN) {
            const float* rowp = xi + (size_t)ih * HIN;
            const float kr0 = (r == 0) ? k0 : (r == 1) ? k3 : k6;
            const float kr1 = (r == 0) ? k1 : (r == 1) ? k4 : k7;
            const float kr2 = (r == 0) ? k2 : (r == 1) ? k5 : k8;
            if (iw0 >= 0)            acc += kr0 * rowp[iw0];
            acc += kr1 * rowp[iw0 + 1];
            if (iw0 + 2 < HIN)       acc += kr2 * rowp[iw0 + 2];
        }
    }
    const float inv  = gamma[c] * __frsqrt_rn(var[c] + EPS_);
    float y = acc * inv + (beta[c] - mean[c] * inv);
    const float a = alpha[c];
    y = (y > 0.f) ? y : a * y;
    out[(((size_t)(b * C_ + c)) * HOUT + oh) * HOUT + ow] = y;
}

// ---------------------------------------------------------------------------
// Kernel 3: Gram matrix  S[b] = (1/64) * P P^T,  P = p2[b] (128 x 4096)
// One 16x16 output tile per wave via v_wmma_f32_16x16x32_bf16, K-steps of 32.
// grid (16, 8), block 128 (4 waves -> 4 tiles); tile = blockIdx.x*4 + waveId
// ---------------------------------------------------------------------------
__global__ void gram_wmma(const float* __restrict__ p2, float* __restrict__ S) {
    const int b    = blockIdx.y;
    const int lane = threadIdx.x & 31;
    const int wid  = threadIdx.x >> 5;
    const int tile = blockIdx.x * 4 + wid;       // 0..63
    const int mt = tile >> 3, nt = tile & 7;

    const float* P = p2 + (size_t)b * C_ * NP_;
    const int arow = mt * 16 + (lane & 15);
    const int brow = nt * 16 + (lane & 15);
    const int kbA  = (lane < 16) ? 0 : 8;        // 16-bit A 16x32 layout
    const int kbB  = (lane < 16) ? 0 : 8;        // 16-bit B 32x16 layout (same k order as A)

    v8f acc = {};
    for (int kt = 0; kt < NP_ / 32; ++kt) {
        v16bf av, bv;
        const float* pa = P + (size_t)arow * NP_ + kt * 32 + kbA;
#pragma unroll
        for (int e = 0; e < 8; ++e) {
            av[e]     = (bf16_t)pa[e];
            av[e + 8] = (bf16_t)pa[16 + e];
        }
        const float* pb = P + (size_t)brow * NP_ + kt * 32 + kbB;
#pragma unroll
        for (int e = 0; e < 8; ++e) { bv[e] = (bf16_t)pb[e]; bv[e + 8] = (bf16_t)pb[16 + e]; }

        acc = __builtin_amdgcn_wmma_f32_16x16x32_bf16(
                  false, av, false, bv, (short)0, acc, false, false);
    }
    // D layout: VGPR j -> row j + 8*(lane>=16), col = lane&15
    const int col   = nt * 16 + (lane & 15);
    const int rbase = mt * 16 + ((lane >> 4) << 3);
    float* Sb = S + (size_t)b * C_ * C_;
#pragma unroll
    for (int j = 0; j < 8; ++j)
        Sb[(size_t)(rbase + j) * C_ + col] = acc[j] * 0.015625f;  // 1/sqrt(4096)
}

// ---------------------------------------------------------------------------
// Kernel 4: row softmax of S[b] (over k) + transpose + bf16:
//   Atr[b][c_out][k] = sim[b,k,c_out]
// One block per batch, one thread per sim row.
// ---------------------------------------------------------------------------
__global__ void softmax_transpose(const float* __restrict__ S,
                                  bf16_t* __restrict__ Atr) {
    const int b = blockIdx.x;
    const int c = threadIdx.x;                   // row of sim
    const float* row = S + (size_t)b * C_ * C_ + (size_t)c * C_;
    float m = -3.4e38f;
    for (int k = 0; k < C_; ++k) m = fmaxf(m, row[k]);
    float s = 0.f;
    for (int k = 0; k < C_; ++k) s += __expf(row[k] - m);
    const float r = 1.f / s;
    bf16_t* Ab = Atr + (size_t)b * C_ * C_;
    for (int k = 0; k < C_; ++k)
        Ab[(size_t)k * C_ + c] = (bf16_t)(__expf(row[k] - m) * r);  // transpose
}

// ---------------------------------------------------------------------------
// Kernel 5: out[b] = Atr[b] (128x128, bf16) @ x[b] (128x65536, f32->bf16)
// Block = 256 threads (8 waves). A^T staged to LDS via the CDNA5 async
// direct-to-LDS path (global_load_async_to_lds_b128 + s_wait_asynccnt).
// Each wave owns a 16-column strip: 4 resident B fragments (x panel read
// once), fully-unrolled sweep of 8 M-tiles -> 32 WMMAs/wave.
// x read exactly once, out written exactly once.  grid (512, 8)
// ---------------------------------------------------------------------------
__global__ void __launch_bounds__(256)
attend_wmma(const float* __restrict__ x, const bf16_t* __restrict__ Atr,
            float* __restrict__ out) {
    const int b    = blockIdx.y;
    const int tid  = threadIdx.x;
    const int lane = tid & 31;
    const int wid  = tid >> 5;

    __shared__ bf16_t ldsA[C_ * C_];             // 32 KB
    {
        // async memory->LDS staging: 2048 x 16B chunks, 8 per thread (GVS mode)
        const bf16_t* gsrc = Atr + (size_t)b * C_ * C_;
        const unsigned ldsbase = (unsigned)(uintptr_t)(void*)ldsA;
#pragma unroll
        for (int k = 0; k < 8; ++k) {
            const int i = tid + k * 256;
            unsigned lds_off = ldsbase + (unsigned)i * 16u;
            unsigned g_off   = (unsigned)i * 16u;
            asm volatile("global_load_async_to_lds_b128 %0, %1, %2 offset:0"
                         :: "v"(lds_off), "v"(g_off), "s"(gsrc)
                         : "memory");
        }
        asm volatile("s_wait_asynccnt 0" ::: "memory");
    }
    __syncthreads();

    const int n0  = (blockIdx.x * 8 + wid) * 16;
    const float* xb = x   + (size_t)b * C_ * N_;
    float*       ob = out + (size_t)b * C_ * N_;

    const int col = n0 + (lane & 15);
    const int kbB = (lane < 16) ? 0 : 8;

    // Load the 128x16 x-panel once: 4 B fragments, f32 -> bf16 in-register
    v16bf Bf[4];
#pragma unroll
    for (int kt = 0; kt < 4; ++kt) {
        const int kb = kt * 32 + kbB;
#pragma unroll
        for (int e = 0; e < 16; ++e)
            Bf[kt][e] = (bf16_t)xb[(size_t)(kb + e + (e & 8)) * N_ + col];
    }

    const int arow = lane & 15;
    const int kbA  = (lane < 16) ? 0 : 8;

#pragma unroll
    for (int mt = 0; mt < 8; ++mt) {
        v8f acc = {};
#pragma unroll
        for (int kt = 0; kt < 4; ++kt) {
            v16bf av;
            const bf16_t* base = &ldsA[(size_t)(mt * 16 + arow) * C_ + kt * 32 + kbA];
            // two 16B-aligned chunks -> b128 ds loads
            const uint4 lo = *(const uint4*)base;
            const uint4 hi = *(const uint4*)(base + 16);
            const bf16_t* plo = (const bf16_t*)&lo;
            const bf16_t* phi = (const bf16_t*)&hi;
#pragma unroll
            for (int e = 0; e < 8; ++e) { av[e] = plo[e]; av[e + 8] = phi[e]; }
            acc = __builtin_amdgcn_wmma_f32_16x16x32_bf16(
                      false, av, false, Bf[kt], (short)0, acc, false, false);
        }
        const int rbase = mt * 16 + ((lane >> 4) << 3);
#pragma unroll
        for (int j = 0; j < 8; ++j)
            ob[(size_t)(rbase + j) * N_ + col] = acc[j];
    }
}

// ---------------------------------------------------------------------------
extern "C" void kernel_launch(void* const* d_in, const int* in_sizes, int n_in,
                              void* d_out, int out_size, void* d_ws, size_t ws_size,
                              hipStream_t stream) {
    const float* x        = (const float*)d_in[0];
    const float* conv1_w  = (const float*)d_in[1];
    const float* bn1_g    = (const float*)d_in[2];
    const float* bn1_b    = (const float*)d_in[3];
    const float* bn1_m    = (const float*)d_in[4];
    const float* bn1_v    = (const float*)d_in[5];
    const float* prelu1_a = (const float*)d_in[6];
    const float* conv2_w  = (const float*)d_in[7];
    const float* bn2_g    = (const float*)d_in[8];
    const float* bn2_b    = (const float*)d_in[9];
    const float* bn2_m    = (const float*)d_in[10];
    const float* bn2_v    = (const float*)d_in[11];
    const float* prelu2_a = (const float*)d_in[12];
    float* out = (float*)d_out;

    // workspace layout
    float*  p1  = (float*)d_ws;                         // 8*128*128*128 f32
    float*  p2  = p1 + (size_t)B_ * C_ * 128 * 128;     // 8*128*64*64  f32
    float*  S   = p2 + (size_t)B_ * C_ * NP_;           // 8*128*128    f32
    bf16_t* Atr = (bf16_t*)(S + (size_t)B_ * C_ * C_);  // 8*128*128    bf16

    // stage 1: 256 -> 128
    dwconv_bn_prelu<256><<<dim3(128, C_, B_), 128, 0, stream>>>(
        x, conv1_w, bn1_g, bn1_b, bn1_m, bn1_v, prelu1_a, p1);
    // stage 2: 128 -> 64
    dwconv_bn_prelu<128><<<dim3(64, C_, B_), 64, 0, stream>>>(
        p1, conv2_w, bn2_g, bn2_b, bn2_m, bn2_v, prelu2_a, p2);
    // Gram matrix (WMMA)
    gram_wmma<<<dim3(16, B_), 128, 0, stream>>>(p2, S);
    // softmax + transpose + bf16
    softmax_transpose<<<dim3(B_), C_, 0, stream>>>(S, Atr);
    // big matmul (WMMA)
    attend_wmma<<<dim3(N_ / 128, B_), 256, 0, stream>>>(x, Atr, out);
}
